// AutoregressiveTokenHead_40621800686275
// MI455X (gfx1250) — matrix operand order repaired and checked by reading, not compiled, not run on hardware
//
#include <hip/hip_runtime.h>
#include <hip/hip_bf16.h>

// ---------------- model constants ----------------
#define BB   16
#define MM   256
#define TT   128
#define DD   512
#define HH   8
#define LL   4
#define VV   32000
#define BOSID 32000
#define BT   (BB * TT)        // 2048 decoder rows
#define BM   (BB * MM)        // 4096 memory rows
#define HD   64               // head dim

typedef __attribute__((ext_vector_type(16))) _Float16 v16h;
typedef __attribute__((ext_vector_type(8)))  float    v8f;

union v8f_u  { v8f  v; float f[8]; };

// ---------------- embedding: x[b,t,:] = tok_emb[shifted_id] + pos_emb[t] --------
__global__ __launch_bounds__(128)
void embed_kernel(const int* __restrict__ targets,
                  const float* __restrict__ tok_emb,
                  const float* __restrict__ pos_emb,
                  float* __restrict__ x)
{
    int bt = blockIdx.x;              // 0..BT-1
    int b = bt / TT, t = bt % TT;
    int id = (t == 0) ? BOSID : targets[b * TT + t - 1];
    const float* te = tok_emb + (size_t)id * DD;
    const float* pe = pos_emb + (size_t)t  * DD;
    float* xr = x + (size_t)bt * DD;
    for (int i = threadIdx.x; i < DD; i += 128)
        xr[i] = te[i] + pe[i];
}

// ---------------- layernorm: one block per row ----------------
__global__ __launch_bounds__(256)
void layernorm_kernel(const float* __restrict__ x,
                      const float* __restrict__ g,
                      const float* __restrict__ b,
                      float* __restrict__ y, int Dn)
{
    __shared__ float red[256];
    int row = blockIdx.x;
    const float* xr = x + (size_t)row * Dn;
    int tid = threadIdx.x;

    float s = 0.f;
    for (int i = tid; i < Dn; i += 256) s += xr[i];
    red[tid] = s; __syncthreads();
    for (int o = 128; o > 0; o >>= 1) {
        if (tid < o) red[tid] += red[tid + o];
        __syncthreads();
    }
    float mean = red[0] / (float)Dn;
    __syncthreads();

    float vs = 0.f;
    for (int i = tid; i < Dn; i += 256) { float d = xr[i] - mean; vs += d * d; }
    red[tid] = vs; __syncthreads();
    for (int o = 128; o > 0; o >>= 1) {
        if (tid < o) red[tid] += red[tid + o];
        __syncthreads();
    }
    float rstd = rsqrtf(red[0] / (float)Dn + 1e-5f);

    float* yr = y + (size_t)row * Dn;
    for (int i = tid; i < Dn; i += 256)
        yr[i] = (xr[i] - mean) * rstd * g[i] + b[i];
}

// ---------------- WMMA GEMM: C = act(A[RxK] @ W[KxN] + bias) (+ res) ------------
// 256 threads = 8 wave32s; macro tile 64(R) x 64(N); K step 32; double-buffered.
// Wave w: row tile tr = w>>1 (0..3), col pair tp = w&1 -> two 16x16 WMMA accs.
// LDS tiles stored in WMMA *fragment order* ([tile][lane][e], 32B/lane) so each
// fragment is one aligned v16h read (ds_load_b128 x2) -- no scalar gather.
#define TILE_R 64
#define TILE_N 64
#define TILE_K 32
#define FRAG_HALVES (4 * 32 * 16)      // 4 tiles x 32 lanes x 16 halves

__global__ __launch_bounds__(256)
void gemm_bias_act_kernel(const float* __restrict__ A,
                          const float* __restrict__ W,
                          const float* __restrict__ bias,
                          const float* __restrict__ res,
                          float* __restrict__ C,
                          int R, int Kdim, int N, int act)
{
    __shared__ __align__(32) _Float16 lsA[2][FRAG_HALVES];   // double-buffered
    __shared__ __align__(32) _Float16 lsB[2][FRAG_HALVES];

    const int tid   = threadIdx.x;
    const int wave  = tid >> 5;
    const int lane  = tid & 31;
    const int tr    = wave >> 1;     // 0..3 row tile
    const int tp    = wave & 1;      // 0..1 col-tile pair
    const int blkN  = blockIdx.x * TILE_N;
    const int blkR  = blockIdx.y * TILE_R;

    // each thread stages 2 float4 chunks of A and 2 of B per K-tile
    const int aIdx0 = tid << 2;            // element index 0..1023
    const int aIdx1 = (tid + 256) << 2;    // 1024..2047
    const int bIdx0 = aIdx0;
    const int bIdx1 = aIdx1;

    auto loadA = [&](int k0, float4& r0, float4& r1) {
        r0 = *(const float4*)&A[(size_t)(blkR + (aIdx0 >> 5)) * Kdim + k0 + (aIdx0 & 31)];
        r1 = *(const float4*)&A[(size_t)(blkR + (aIdx1 >> 5)) * Kdim + k0 + (aIdx1 & 31)];
    };
    auto loadB = [&](int k0, float4& r0, float4& r1) {
        r0 = *(const float4*)&W[(size_t)(k0 + (bIdx0 >> 6)) * N + blkN + (bIdx0 & 63)];
        r1 = *(const float4*)&W[(size_t)(k0 + (bIdx1 >> 6)) * N + blkN + (bIdx1 & 63)];
    };
    // inverse of ISA A-frag layout: lane=(r&15)+16*((c>>3)&1), e=(c&7)|((c>>1)&8)
    auto storeA = [&](int p, const float4& v4, int idx) {
        int r = idx >> 5, cb = idx & 31;
        int trr = r >> 4, row = r & 15;
        float vv[4] = { v4.x, v4.y, v4.z, v4.w };
        #pragma unroll
        for (int q = 0; q < 4; ++q) {
            int c = cb + q;
            int la = row + (((c >> 3) & 1) << 4);
            int e  = (c & 7) | ((c >> 1) & 8);
            lsA[p][((trr * 32 + la) << 4) + e] = (_Float16)vv[q];
        }
    };
    // inverse of ISA B-frag layout: lane=(c&15)+16*((k>>3)&1), e=(k&7)|((k>>1)&8)
    auto storeB = [&](int p, const float4& v4, int idx) {
        int kk = idx >> 6, cb = idx & 63;
        int kh = (kk >> 3) & 1, e = (kk & 7) | ((kk >> 1) & 8);
        float vv[4] = { v4.x, v4.y, v4.z, v4.w };
        #pragma unroll
        for (int q = 0; q < 4; ++q) {
            int c = cb + q;
            int la = (c & 15) + (kh << 4);
            lsB[p][(((c >> 4) * 32 + la) << 4) + e] = (_Float16)vv[q];
        }
    };

    v8f_u acc0, acc1;
    #pragma unroll
    for (int i = 0; i < 8; ++i) { acc0.f[i] = 0.f; acc1.f[i] = 0.f; }

    const int l16   = lane & 15;
    const int khalf = lane >> 4;

    // ---- prologue: stage K-tile 0 into buffer 0 ----
    float4 a0, a1, b0, b1;
    loadA(0, a0, a1);
    loadB(0, b0, b1);
    storeA(0, a0, aIdx0); storeA(0, a1, aIdx1);
    storeB(0, b0, bIdx0); storeB(0, b1, bIdx1);
    __syncthreads();

    int p = 0;
    for (int k0 = 0; k0 < Kdim; k0 += TILE_K) {
        const bool hasNext = (k0 + TILE_K) < Kdim;
        // fetch next tile into registers while this tile computes
        if (hasNext) {
            loadA(k0 + TILE_K, a0, a1);
            loadB(k0 + TILE_K, b0, b1);
        }
        // hint HBM two tiles ahead (global_prefetch_b8)
        if (k0 + 2 * TILE_K < Kdim) {
            __builtin_prefetch(&A[(size_t)(blkR + (tid >> 2)) * Kdim +
                                  (k0 + 2 * TILE_K) + ((tid & 3) << 3)], 0, 1);
            __builtin_prefetch(&W[(size_t)(k0 + 2 * TILE_K + (tid >> 3)) * N +
                                  blkN + ((tid & 7) << 3)], 0, 1);
        }

        // ---- fragments: aligned v16h loads (ds_load_b128 x2 each) ----
        const v16h af = *(const v16h*)(&lsA[p][(tr * 32 + lane) << 4]);
        const v16h bf0 = *(const v16h*)(&lsB[p][((tp * 2 + 0) * 32 + lane) << 4]);
        const v16h bf1 = *(const v16h*)(&lsB[p][((tp * 2 + 1) * 32 + lane) << 4]);

        acc0.v = __builtin_amdgcn_wmma_f32_16x16x32_f16(
            false, af, false, bf0, (short)0, acc0.v, false, false);
        acc1.v = __builtin_amdgcn_wmma_f32_16x16x32_f16(
            false, af, false, bf1, (short)0, acc1.v, false, false);

        // stage next tile into the other buffer
        if (hasNext) {
            storeA(p ^ 1, a0, aIdx0); storeA(p ^ 1, a1, aIdx1);
            storeB(p ^ 1, b0, bIdx0); storeB(p ^ 1, b1, bIdx1);
        }
        __syncthreads();
        p ^= 1;
    }

    // ---- epilogue: C/D layout -> VGPR r: row = r + 8*(lane>>4), col = lane&15
    const int rowBase = blkR + tr * 16 + khalf * 8;
    #pragma unroll
    for (int half = 0; half < 2; ++half) {
        const float* af = half ? acc1.f : acc0.f;
        const int col = blkN + (tp * 2 + half) * 16 + l16;
        const float bv = bias ? bias[col] : 0.f;
        #pragma unroll
        for (int r = 0; r < 8; ++r) {
            int row = rowBase + r;
            float v = af[r] + bv;
            if (act == 1)  // exact-erf GELU
                v = 0.5f * v * (1.f + erff(v * 0.70710678118654752f));
            if (res) v += res[(size_t)row * N + col];
            C[(size_t)row * N + col] = v;
        }
    }
}

// ---------------- attention: one block per (b, h, query t) ----------------------
__global__ __launch_bounds__(128)
void attention_kernel(const float* __restrict__ Qb, int qstride,
                      const float* __restrict__ Kb, int kstride,
                      const float* __restrict__ Vb, int vstride,
                      float* __restrict__ O,
                      int Tq, int Tk, int causal)
{
    __shared__ float qs[HD];
    __shared__ float sbuf[256];   // Tk <= 256
    __shared__ float red[128];

    const int bid = blockIdx.x;
    const int t = bid % Tq;
    const int h = (bid / Tq) % HH;
    const int b = bid / (Tq * HH);
    const int tid = threadIdx.x;

    const float* qrow = Qb + (size_t)(b * Tq + t) * qstride + h * HD;
    if (tid < HD) qs[tid] = qrow[tid];
    __syncthreads();

    const float scale = 0.125f;   // 1/sqrt(64)
    float lmax = -1e30f;
    for (int j = tid; j < Tk; j += 128) {
        const float* krow = Kb + (size_t)(b * Tk + j) * kstride + h * HD;
        float s = 0.f;
        #pragma unroll 8
        for (int d = 0; d < HD; ++d) s += qs[d] * krow[d];
        s *= scale;
        if (causal && j > t) s = -1e9f;
        sbuf[j] = s;
        lmax = fmaxf(lmax, s);
    }
    red[tid] = lmax; __syncthreads();
    for (int o = 64; o > 0; o >>= 1) {
        if (tid < o) red[tid] = fmaxf(red[tid], red[tid + o]);
        __syncthreads();
    }
    const float mx = red[0];
    __syncthreads();

    float lsum = 0.f;
    for (int j = tid; j < Tk; j += 128) {
        float pj = __expf(sbuf[j] - mx);
        sbuf[j] = pj;
        lsum += pj;
    }
    red[tid] = lsum; __syncthreads();
    for (int o = 64; o > 0; o >>= 1) {
        if (tid < o) red[tid] += red[tid + o];
        __syncthreads();
    }
    const float inv = 1.f / red[0];
    __syncthreads();

    if (tid < HD) {
        float o = 0.f;
        for (int j = 0; j < Tk; ++j)
            o += sbuf[j] * Vb[(size_t)(b * Tk + j) * vstride + h * HD + tid];
        O[(size_t)(b * Tq + t) * DD + h * HD + tid] = o * inv;
    }
}

// ---------------- host-side orchestration ----------------
static void launch_gemm(const float* A, const float* W, const float* bias,
                        const float* res, float* C, int R, int K, int N,
                        int act, hipStream_t s)
{
    dim3 grid(N / TILE_N, R / TILE_R);
    gemm_bias_act_kernel<<<grid, 256, 0, s>>>(A, W, bias, res, C, R, K, N, act);
}

extern "C" void kernel_launch(void* const* d_in, const int* in_sizes, int n_in,
                              void* d_out, int out_size, void* d_ws, size_t ws_size,
                              hipStream_t stream)
{
    (void)in_sizes; (void)n_in; (void)out_size; (void)ws_size;

    const float* memory   = (const float*)d_in[0];
    const int*   targets  = (const int*  )d_in[1];
    const float* tok_emb  = (const float*)d_in[2];
    const float* pos_emb  = (const float*)d_in[3];
    const float* sa_qkv_w = (const float*)d_in[4];
    const float* sa_qkv_b = (const float*)d_in[5];
    const float* sa_out_w = (const float*)d_in[6];
    const float* sa_out_b = (const float*)d_in[7];
    const float* ca_q_w   = (const float*)d_in[8];
    const float* ca_q_b   = (const float*)d_in[9];
    const float* ca_kv_w  = (const float*)d_in[10];
    const float* ca_kv_b  = (const float*)d_in[11];
    const float* ca_out_w = (const float*)d_in[12];
    const float* ca_out_b = (const float*)d_in[13];
    const float* ffn1_w   = (const float*)d_in[14];
    const float* ffn1_b   = (const float*)d_in[15];
    const float* ffn2_w   = (const float*)d_in[16];
    const float* ffn2_b   = (const float*)d_in[17];
    const float* ln1_g    = (const float*)d_in[18];
    const float* ln1_b    = (const float*)d_in[19];
    const float* ln2_g    = (const float*)d_in[20];
    const float* ln2_b    = (const float*)d_in[21];
    const float* ln3_g    = (const float*)d_in[22];
    const float* ln3_b    = (const float*)d_in[23];
    const float* normf_g  = (const float*)d_in[24];
    const float* normf_b  = (const float*)d_in[25];
    const float* out_w    = (const float*)d_in[26];
    const float* out_b    = (const float*)d_in[27];
    float* logits = (float*)d_out;

    // workspace carve-out (floats): ~59 MB total
    float* ws = (float*)d_ws;
    size_t o = 0;
    float* x    = ws + o;  o += (size_t)BT * DD;          // activations  [2048,512]
    float* hbuf = ws + o;  o += (size_t)BT * DD;          // ln output    [2048,512]
    float* qkv  = ws + o;  o += (size_t)BT * 3 * DD;      // qkv / ca_q   [2048,1536]
    float* attn = ws + o;  o += (size_t)BT * DD;          // attn merged  [2048,512]
    float* kvm  = ws + o;  o += (size_t)BM * 2 * DD;      // memory k|v   [4096,1024]
    float* ffn  = ws + o;  o += (size_t)BT * 4 * DD;      // ffn hidden   [2048,2048]

    embed_kernel<<<BT, 128, 0, stream>>>(targets, tok_emb, pos_emb, x);

    for (int l = 0; l < LL; ++l) {
        // ---- self-attention ----
        layernorm_kernel<<<BT, 256, 0, stream>>>(x, ln1_g + l * DD, ln1_b + l * DD, hbuf, DD);
        launch_gemm(hbuf, sa_qkv_w + (size_t)l * DD * 3 * DD, sa_qkv_b + (size_t)l * 3 * DD,
                    nullptr, qkv, BT, DD, 3 * DD, 0, stream);
        attention_kernel<<<BB * HH * TT, 128, 0, stream>>>(
            qkv,          3 * DD,
            qkv + DD,     3 * DD,
            qkv + 2 * DD, 3 * DD,
            attn, TT, TT, /*causal=*/1);
        launch_gemm(attn, sa_out_w + (size_t)l * DD * DD, sa_out_b + (size_t)l * DD,
                    x, x, BT, DD, DD, 0, stream);

        // ---- cross-attention ----
        layernorm_kernel<<<BT, 256, 0, stream>>>(x, ln2_g + l * DD, ln2_b + l * DD, hbuf, DD);
        launch_gemm(hbuf, ca_q_w + (size_t)l * DD * DD, ca_q_b + (size_t)l * DD,
                    nullptr, qkv, BT, DD, DD, 0, stream);
        launch_gemm(memory, ca_kv_w + (size_t)l * DD * 2 * DD, ca_kv_b + (size_t)l * 2 * DD,
                    nullptr, kvm, BM, DD, 2 * DD, 0, stream);
        attention_kernel<<<BB * HH * TT, 128, 0, stream>>>(
            qkv,      DD,
            kvm,      2 * DD,
            kvm + DD, 2 * DD,
            attn, TT, MM, /*causal=*/0);
        launch_gemm(attn, ca_out_w + (size_t)l * DD * DD, ca_out_b + (size_t)l * DD,
                    x, x, BT, DD, DD, 0, stream);

        // ---- FFN (exact-erf GELU) ----
        layernorm_kernel<<<BT, 256, 0, stream>>>(x, ln3_g + l * DD, ln3_b + l * DD, hbuf, DD);
        launch_gemm(hbuf, ffn1_w + (size_t)l * DD * 4 * DD, ffn1_b + (size_t)l * 4 * DD,
                    nullptr, ffn, BT, DD, 4 * DD, 1, stream);
        launch_gemm(ffn, ffn2_w + (size_t)l * 4 * DD * DD, ffn2_b + (size_t)l * DD,
                    x, x, BT, 4 * DD, DD, 0, stream);
    }

    // final norm + vocab head (HBM-bound: streams 65 MB of out_w)
    layernorm_kernel<<<BT, 256, 0, stream>>>(x, normf_g, normf_b, hbuf, DD);
    launch_gemm(hbuf, out_w, out_b, nullptr, logits, BT, DD, VV, 0, stream);
}
